// MobileNetBlock_66194035966283
// MI455X (gfx1250) — 8-bit matrix operand order repaired and checked by reading, compiled, not run on hardware
//
#include <hip/hip_runtime.h>
#include <hip/hip_bf16.h>

#define B_N    16
#define C_IN   128
#define E_CH   768
#define H_IMG  56
#define W_IMG  56
#define HW_IMG 3136
#define P_TOT  50176
#define EPS_BN 1e-5f

typedef __attribute__((ext_vector_type(8))) int v8i;

__device__ __forceinline__ signed char sgn8(float v) {
  return (signed char)((v > 0.f) ? 1 : ((v < 0.f) ? -1 : 0));
}
__device__ __forceinline__ int isgn(float v) {
  return (v > 0.f) ? 1 : ((v < 0.f) ? -1 : 0);
}
__device__ __forceinline__ float fsgn(float v) {
  return (v > 0.f) ? 1.f : ((v < 0.f) ? -1.f : 0.f);
}
__device__ __forceinline__ float htanh(float v) {
  return fminf(1.f, fmaxf(-1.f, v));
}

__global__ void k_zero(float* __restrict__ p, int n) {
  int i = blockIdx.x * blockDim.x + threadIdx.x;
  if (i < n) p[i] = 0.f;
}

__global__ void k_binarize_x(const float* __restrict__ x, signed char* __restrict__ A1) {
  int idx = blockIdx.x * blockDim.x + threadIdx.x;
  int p  = idx >> 5;
  int c4 = (idx & 31) * 4;
  if (p >= P_TOT) return;
  int b = p / HW_IMG, hw = p % HW_IMG;
  const float* xp = x + ((size_t)b * C_IN + c4) * HW_IMG + hw;
  char4 o;
  o.x = sgn8(xp[0]);
  o.y = sgn8(xp[HW_IMG]);
  o.z = sgn8(xp[2 * HW_IMG]);
  o.w = sgn8(xp[3 * HW_IMG]);
  *(char4*)(A1 + (size_t)p * C_IN + c4) = o;
}

__global__ void k_pack_weights(const float* __restrict__ w1, const float* __restrict__ w3,
                               const float* __restrict__ w2,
                               signed char* __restrict__ B1, signed char* __restrict__ B3,
                               signed char* __restrict__ SW2) {
  int i = blockIdx.x * blockDim.x + threadIdx.x;
  if (i < E_CH * C_IN) B1[i] = sgn8(w1[i]);
  if (i < C_IN * E_CH) B3[i] = sgn8(w3[i]);
  if (i < E_CH * 9)    SW2[i] = sgn8(w2[i]);
}

__global__ void k_bn_finalize(const float* __restrict__ sum, const float* __restrict__ sumsq,
                              const float* __restrict__ gamma, const float* __restrict__ beta,
                              float invcnt, int n, float* __restrict__ a, float* __restrict__ c) {
  int i = blockIdx.x * blockDim.x + threadIdx.x;
  if (i >= n) return;
  float m = sum[i] * invcnt;
  float v = sumsq[i] * invcnt - m * m;
  float s = gamma[i] * rsqrtf(v + EPS_BN);
  a[i] = s;
  c[i] = beta[i] - s * m;
}

__global__ void __launch_bounds__(256)
k_gemm1(const signed char* __restrict__ A1, const signed char* __restrict__ B1,
        short* __restrict__ Y1, float* __restrict__ sum1, float* __restrict__ sq1) {
  const int lane = threadIdx.x & 31;
  const int wave = threadIdx.x >> 5;
  const int m0 = blockIdx.x * 128 + wave * 16;
  const int n0 = blockIdx.y * 64;
  const int g   = lane >> 4;
  const int l15 = lane & 15;

  v8i acc[4] = {};
  const signed char* arowb = A1 + (size_t)(m0 + l15) * C_IN;
  __builtin_prefetch(arowb + 64, 0, 1);

#pragma unroll
  for (int ka = 0; ka < 128; ka += 64) {
    const int* arow = (const int*)(arowb + ka);
    v8i a;
    int2 t;
    t = *(const int2*)(arow + 0  + 4 * g); a[0] = t.x; a[1] = t.y;
    t = *(const int2*)(arow + 2  + 4 * g); a[2] = t.x; a[3] = t.y;
    t = *(const int2*)(arow + 8  + 4 * g); a[4] = t.x; a[5] = t.y;
    t = *(const int2*)(arow + 10 + 4 * g); a[6] = t.x; a[7] = t.y;
#pragma unroll
    for (int ns = 0; ns < 4; ++ns) {
      const int* bcol = (const int*)(B1 + (size_t)(n0 + ns * 16 + l15) * C_IN + ka);
      int4 q0 = *(const int4*)(bcol + 4 * g);
      int4 q1 = *(const int4*)(bcol + 8 + 4 * g);
      v8i b;
      b[0] = q0.x; b[1] = q0.y; b[2] = q0.z; b[3] = q0.w;
      b[4] = q1.x; b[5] = q1.y; b[6] = q1.z; b[7] = q1.w;
      acc[ns] = __builtin_amdgcn_wmma_i32_16x16x64_iu8(true, a, true, b, acc[ns], false, false);
    }
  }
#pragma unroll
  for (int ns = 0; ns < 4; ++ns) {
    int n = n0 + ns * 16 + l15;
    float ls = 0.f, lq = 0.f;
#pragma unroll
    for (int r = 0; r < 8; ++r) {
      int m = m0 + r + 8 * g;
      int v = acc[ns][r];
      Y1[(size_t)m * E_CH + n] = (short)v;
      float f = (float)v;
      ls += f; lq += f * f;
    }
    atomicAdd(&sum1[n], ls);
    atomicAdd(&sq1[n],  lq);
  }
}

__global__ void __launch_bounds__(192)
k_dw(const short* __restrict__ Y1, const signed char* __restrict__ SW2,
     const float* __restrict__ a1, const float* __restrict__ c1,
     signed char* __restrict__ D, float* __restrict__ sum2, float* __restrict__ sq2) {
  int bh = blockIdx.x;
  int b = bh / H_IMG, h = bh % H_IMG;
  int e4 = threadIdx.x * 4;
  float la[4], lc[4];
  int sw[4][9];
#pragma unroll
  for (int u = 0; u < 4; ++u) {
    la[u] = a1[e4 + u]; lc[u] = c1[e4 + u];
#pragma unroll
    for (int t = 0; t < 9; ++t) sw[u][t] = (int)SW2[(e4 + u) * 9 + t];
  }
  float ls[4] = {0, 0, 0, 0}, lq[4] = {0, 0, 0, 0};
  for (int w = 0; w < W_IMG; ++w) {
    int acc[4] = {0, 0, 0, 0};
#pragma unroll
    for (int ky = 0; ky < 3; ++ky) {
      int hh = h + ky - 1;
      if (hh < 0 || hh >= H_IMG) continue;
#pragma unroll
      for (int kx = 0; kx < 3; ++kx) {
        int ww = w + kx - 1;
        if (ww < 0 || ww >= W_IMG) continue;
        size_t p = (size_t)b * HW_IMG + hh * W_IMG + ww;
        short4 yv = *(const short4*)(Y1 + p * E_CH + e4);
        acc[0] += isgn(la[0] * (float)yv.x + lc[0]) * sw[0][ky * 3 + kx];
        acc[1] += isgn(la[1] * (float)yv.y + lc[1]) * sw[1][ky * 3 + kx];
        acc[2] += isgn(la[2] * (float)yv.z + lc[2]) * sw[2][ky * 3 + kx];
        acc[3] += isgn(la[3] * (float)yv.w + lc[3]) * sw[3][ky * 3 + kx];
      }
    }
    size_t p = (size_t)b * HW_IMG + h * W_IMG + w;
    char4 dv;
    dv.x = (signed char)acc[0]; dv.y = (signed char)acc[1];
    dv.z = (signed char)acc[2]; dv.w = (signed char)acc[3];
    *(char4*)(D + p * E_CH + e4) = dv;
#pragma unroll
    for (int u = 0; u < 4; ++u) {
      float f = (float)acc[u];
      ls[u] += f; lq[u] += f * f;
    }
  }
#pragma unroll
  for (int u = 0; u < 4; ++u) {
    atomicAdd(&sum2[e4 + u], ls[u]);
    atomicAdd(&sq2[e4 + u],  lq[u]);
  }
}

__global__ void __launch_bounds__(256)
k_gemm2(const signed char* __restrict__ D, const signed char* __restrict__ B3,
        const float* __restrict__ a2, const float* __restrict__ c2,
        int* __restrict__ Y3, float* __restrict__ sum3, float* __restrict__ sq3) {
  __shared__ float sa[E_CH];
  __shared__ float sc[E_CH];
  for (int i = threadIdx.x; i < E_CH; i += 256) { sa[i] = a2[i]; sc[i] = c2[i]; }
  __syncthreads();

  const int lane = threadIdx.x & 31;
  const int wave = threadIdx.x >> 5;
  const int m0 = blockIdx.x * 128 + wave * 16;
  const int n0 = blockIdx.y * 64;
  const int g   = lane >> 4;
  const int l15 = lane & 15;

  v8i acc[4] = {};
  const signed char* drow = D + (size_t)(m0 + l15) * E_CH;

  for (int kc = 0; kc < 12; ++kc) {
    int ka = kc * 64;
    __builtin_prefetch(drow + ka + 64, 0, 1);
    v8i a;
#pragma unroll
    for (int v = 0; v < 8; ++v) {
      int dw = (v >> 2) * 8 + 4 * g + (v & 3);
      int k4 = ka + dw * 4;
      char4 dv = *(const char4*)(drow + ka + dw * 4);
      int s0 = isgn(sa[k4 + 0] * (float)dv.x + sc[k4 + 0]);
      int s1 = isgn(sa[k4 + 1] * (float)dv.y + sc[k4 + 1]);
      int s2 = isgn(sa[k4 + 2] * (float)dv.z + sc[k4 + 2]);
      int s3 = isgn(sa[k4 + 3] * (float)dv.w + sc[k4 + 3]);
      a[v] = (s0 & 0xFF) | ((s1 & 0xFF) << 8) | ((s2 & 0xFF) << 16) | (s3 << 24);
    }
#pragma unroll
    for (int ns = 0; ns < 4; ++ns) {
      const int* bcol = (const int*)(B3 + (size_t)(n0 + ns * 16 + l15) * E_CH + ka);
      int4 q0 = *(const int4*)(bcol + 4 * g);
      int4 q1 = *(const int4*)(bcol + 8 + 4 * g);
      v8i b;
      b[0] = q0.x; b[1] = q0.y; b[2] = q0.z; b[3] = q0.w;
      b[4] = q1.x; b[5] = q1.y; b[6] = q1.z; b[7] = q1.w;
      acc[ns] = __builtin_amdgcn_wmma_i32_16x16x64_iu8(true, a, true, b, acc[ns], false, false);
    }
  }
#pragma unroll
  for (int ns = 0; ns < 4; ++ns) {
    int n = n0 + ns * 16 + l15;
    float ls = 0.f, lq = 0.f;
#pragma unroll
    for (int r = 0; r < 8; ++r) {
      int m = m0 + r + 8 * g;
      int v = acc[ns][r];
      Y3[(size_t)m * C_IN + n] = v;
      float f = (float)v;
      ls += f; lq += f * f;
    }
    atomicAdd(&sum3[n], ls);
    atomicAdd(&sq3[n],  lq);
  }
}

__global__ void __launch_bounds__(128)
k_pool(const int* __restrict__ Y3, const float* __restrict__ a3, const float* __restrict__ c3,
       float* __restrict__ pool) {
  int b = blockIdx.x;
  int c = threadIdx.x;
  float aa = a3[c], cc = c3[c];
  float s = 0.f;
  for (int p = 0; p < HW_IMG; ++p) {
    int v = Y3[((size_t)b * HW_IMG + p) * C_IN + c];
    s += htanh(aa * (float)v + cc);
  }
  pool[b * C_IN + c] = s * (1.f / HW_IMG);
}

__global__ void __launch_bounds__(256)
k_se(const float* __restrict__ pool,
     const float* __restrict__ g_se1, const float* __restrict__ b_se1,
     const float* __restrict__ w_fc2, const float* __restrict__ bias_fc2,
     const float* __restrict__ g_se2, const float* __restrict__ b_se2,
     const float* __restrict__ w_fc3, const float* __restrict__ bias_fc3,
     const float* __restrict__ g_se3, const float* __restrict__ b_se3,
     float* __restrict__ gate) {
  __shared__ float sm[B_N * C_IN];
  __shared__ float sb1[B_N * C_IN];
  __shared__ float h2[B_N * 16];
  __shared__ float s2[B_N * 16];
  __shared__ float o3[B_N * C_IN];
  int t = threadIdx.x;
  for (int i = t; i < B_N * C_IN; i += 256) sm[i] = pool[i];
  __syncthreads();
  if (t < C_IN) {
    float m = 0.f;
    for (int b = 0; b < B_N; ++b) m += sm[b * C_IN + t];
    m *= (1.f / B_N);
    float v = 0.f;
    for (int b = 0; b < B_N; ++b) { float d = sm[b * C_IN + t] - m; v += d * d; }
    v *= (1.f / B_N);
    float s = g_se1[t] * rsqrtf(v + EPS_BN);
    float sh = b_se1[t] - s * m;
    for (int b = 0; b < B_N; ++b)
      sb1[b * C_IN + t] = fsgn(s * sm[b * C_IN + t] + sh);
  }
  __syncthreads();
  {
    int b = t >> 4, j = t & 15;
    float acc = bias_fc2[j];
    for (int c = 0; c < C_IN; ++c) acc += sb1[b * C_IN + c] * fsgn(w_fc2[j * C_IN + c]);
    h2[b * 16 + j] = acc;
  }
  __syncthreads();
  if (t < 16) {
    float m = 0.f;
    for (int b = 0; b < B_N; ++b) m += h2[b * 16 + t];
    m *= (1.f / B_N);
    float v = 0.f;
    for (int b = 0; b < B_N; ++b) { float d = h2[b * 16 + t] - m; v += d * d; }
    v *= (1.f / B_N);
    float s = g_se2[t] * rsqrtf(v + EPS_BN);
    float sh = b_se2[t] - s * m;
    for (int b = 0; b < B_N; ++b) s2[b * 16 + t] = fsgn(s * h2[b * 16 + t] + sh);
  }
  __syncthreads();
  for (int i = t; i < B_N * C_IN; i += 256) {
    int b = i >> 7, c = i & 127;
    float acc = bias_fc3[c];
    for (int j = 0; j < 16; ++j) acc += s2[b * 16 + j] * fsgn(w_fc3[c * 16 + j]);
    o3[i] = acc;
  }
  __syncthreads();
  if (t < C_IN) {
    float m = 0.f;
    for (int b = 0; b < B_N; ++b) m += o3[b * C_IN + t];
    m *= (1.f / B_N);
    float v = 0.f;
    for (int b = 0; b < B_N; ++b) { float d = o3[b * C_IN + t] - m; v += d * d; }
    v *= (1.f / B_N);
    float s = g_se3[t] * rsqrtf(v + EPS_BN);
    float sh = b_se3[t] - s * m;
    for (int b = 0; b < B_N; ++b) {
      float z = s * o3[b * C_IN + t] + sh;
      gate[b * C_IN + t] = 1.f / (1.f + __expf(-z));
    }
  }
}

__global__ void __launch_bounds__(256)
k_zres(const int* __restrict__ Y3, const float* __restrict__ a3, const float* __restrict__ c3,
       const float* __restrict__ gate, const float* __restrict__ x,
       float* __restrict__ Z, float* __restrict__ sumr, float* __restrict__ sqr) {
  __shared__ float ssum[C_IN];
  __shared__ float ssq[C_IN];
  if (threadIdx.x < C_IN) { ssum[threadIdx.x] = 0.f; ssq[threadIdx.x] = 0.f; }
  __syncthreads();
  int idx = blockIdx.x * 256 + threadIdx.x;
  int p  = idx >> 5;
  int c4 = (idx & 31) * 4;
  if (p < P_TOT) {
    int b = p / HW_IMG, hw = p % HW_IMG;
    int4 yv = *(const int4*)(Y3 + (size_t)p * C_IN + c4);
    int ys[4] = {yv.x, yv.y, yv.z, yv.w};
#pragma unroll
    for (int u = 0; u < 4; ++u) {
      int c = c4 + u;
      float o = htanh(a3[c] * (float)ys[u] + c3[c]);
      float z = o * gate[b * C_IN + c] + x[((size_t)b * C_IN + c) * HW_IMG + hw];
      Z[(size_t)p * C_IN + c] = z;
      atomicAdd(&ssum[c], z);
      atomicAdd(&ssq[c],  z * z);
    }
  }
  __syncthreads();
  if (threadIdx.x < C_IN) {
    atomicAdd(&sumr[threadIdx.x], ssum[threadIdx.x]);
    atomicAdd(&sqr[threadIdx.x],  ssq[threadIdx.x]);
  }
}

__global__ void k_out(const float* __restrict__ Z, const float* __restrict__ ar,
                      const float* __restrict__ cr, float* __restrict__ out) {
  size_t i = (size_t)blockIdx.x * 256 + threadIdx.x;
  if (i >= (size_t)B_N * C_IN * HW_IMG) return;
  int hw = (int)(i % HW_IMG);
  size_t bc = i / HW_IMG;
  int c = (int)(bc % C_IN);
  int b = (int)(bc / C_IN);
  float z = Z[((size_t)b * HW_IMG + hw) * C_IN + c];
  out[i] = htanh(ar[c] * z + cr[c]);
}

extern "C" void kernel_launch(void* const* d_in, const int* in_sizes, int n_in,
                              void* d_out, int out_size, void* d_ws, size_t ws_size,
                              hipStream_t stream) {
  const float* x        = (const float*)d_in[0];
  const float* w1       = (const float*)d_in[1];
  const float* g1       = (const float*)d_in[2];
  const float* b1       = (const float*)d_in[3];
  const float* w2       = (const float*)d_in[4];
  const float* g2       = (const float*)d_in[5];
  const float* b2       = (const float*)d_in[6];
  const float* w3       = (const float*)d_in[7];
  const float* g3       = (const float*)d_in[8];
  const float* b3       = (const float*)d_in[9];
  const float* g_res    = (const float*)d_in[10];
  const float* b_res    = (const float*)d_in[11];
  const float* g_se1    = (const float*)d_in[12];
  const float* b_se1    = (const float*)d_in[13];
  const float* w_fc2    = (const float*)d_in[14];
  const float* bias_fc2 = (const float*)d_in[15];
  const float* g_se2    = (const float*)d_in[16];
  const float* b_se2    = (const float*)d_in[17];
  const float* w_fc3    = (const float*)d_in[18];
  const float* bias_fc3 = (const float*)d_in[19];
  const float* g_se3    = (const float*)d_in[20];
  const float* b_se3    = (const float*)d_in[21];

  char* ws = (char*)d_ws;
  size_t off = 0;
  auto carve = [&](size_t bytes) { size_t r = off; off = (off + bytes + 255) & ~(size_t)255; return r; };

  float* stats = (float*)(ws + carve((4 * E_CH + 4 * C_IN + B_N * C_IN) * sizeof(float)));
  float* sum1 = stats;            float* sq1 = sum1 + E_CH;
  float* sum2 = sq1 + E_CH;       float* sq2 = sum2 + E_CH;
  float* sum3 = sq2 + E_CH;       float* sq3 = sum3 + C_IN;
  float* sumr = sq3 + C_IN;       float* sqr = sumr + C_IN;
  float* pool = sqr + C_IN;
  const int n_stats = 4 * E_CH + 4 * C_IN + B_N * C_IN;

  float* params = (float*)(ws + carve((4 * E_CH + 4 * C_IN + B_N * C_IN) * sizeof(float)));
  float* a1c = params;            float* c1c = a1c + E_CH;
  float* a2c = c1c + E_CH;        float* c2c = a2c + E_CH;
  float* a3c = c2c + E_CH;        float* c3c = a3c + C_IN;
  float* arc = c3c + C_IN;        float* crc = arc + C_IN;
  float* gate = crc + C_IN;

  signed char* A1  = (signed char*)(ws + carve((size_t)P_TOT * C_IN));
  signed char* B1  = (signed char*)(ws + carve((size_t)E_CH * C_IN));
  signed char* B3  = (signed char*)(ws + carve((size_t)C_IN * E_CH));
  signed char* SW2 = (signed char*)(ws + carve((size_t)E_CH * 9));
  size_t y1_off = carve((size_t)P_TOT * E_CH * sizeof(short));
  short* Y1 = (short*)(ws + y1_off);
  float* Z  = (float*)(ws + y1_off);
  signed char* Dq = (signed char*)(ws + carve((size_t)P_TOT * E_CH));
  int* Y3 = (int*)(ws + carve((size_t)P_TOT * C_IN * sizeof(int)));

  const float invP = 1.f / (float)P_TOT;

  k_zero<<<(n_stats + 255) / 256, 256, 0, stream>>>(stats, n_stats);
  k_binarize_x<<<(P_TOT * 32 + 255) / 256, 256, 0, stream>>>(x, A1);
  k_pack_weights<<<(E_CH * C_IN + 255) / 256, 256, 0, stream>>>(w1, w3, w2, B1, B3, SW2);

  k_gemm1<<<dim3(P_TOT / 128, E_CH / 64), 256, 0, stream>>>(A1, B1, Y1, sum1, sq1);
  k_bn_finalize<<<(E_CH + 255) / 256, 256, 0, stream>>>(sum1, sq1, g1, b1, invP, E_CH, a1c, c1c);

  k_dw<<<B_N * H_IMG, 192, 0, stream>>>(Y1, SW2, a1c, c1c, Dq, sum2, sq2);
  k_bn_finalize<<<(E_CH + 255) / 256, 256, 0, stream>>>(sum2, sq2, g2, b2, invP, E_CH, a2c, c2c);

  k_gemm2<<<dim3(P_TOT / 128, C_IN / 64), 256, 0, stream>>>(Dq, B3, a2c, c2c, Y3, sum3, sq3);
  k_bn_finalize<<<1, 128, 0, stream>>>(sum3, sq3, g3, b3, invP, C_IN, a3c, c3c);

  k_pool<<<B_N, 128, 0, stream>>>(Y3, a3c, c3c, pool);
  k_se<<<1, 256, 0, stream>>>(pool, g_se1, b_se1, w_fc2, bias_fc2, g_se2, b_se2,
                              w_fc3, bias_fc3, g_se3, b_se3, gate);

  k_zres<<<(P_TOT * 32 + 255) / 256, 256, 0, stream>>>(Y3, a3c, c3c, gate, x, Z, sumr, sqr);
  k_bn_finalize<<<1, 128, 0, stream>>>(sumr, sqr, g_res, b_res, invP, C_IN, arc, crc);

  k_out<<<(int)(((size_t)B_N * C_IN * HW_IMG + 255) / 256), 256, 0, stream>>>(Z, arc, crc, (float*)d_out);
}
